// NN_O_61993557950504
// MI455X (gfx1250) — matrix operand order repaired and checked by reading, not compiled, not run on hardware
//
#include <hip/hip_runtime.h>

typedef __attribute__((ext_vector_type(16))) _Float16 v16h;
typedef __attribute__((ext_vector_type(8)))  float    v8f;

#define B_   16
#define P_   64
#define T_   20
#define O_   32
#define DIM_ 128
#define NROWS (B_*P_*T_*O_)          /* 655360 */
#define EPS_ 1e-5f
#define NEG_SLOPE_ 0.1f
#define SENSING_ 2.0f
#define HALF_ANGLE (0.5f * 0.8333f * 3.1415926f)

__device__ __forceinline__ float leaky_(float x) { return x >= 0.f ? x : NEG_SLOPE_ * x; }

__device__ __forceinline__ unsigned int pack2_(float a, float b) {
    union { _Float16 h[2]; unsigned int u; } c;
    c.h[0] = (_Float16)a; c.h[1] = (_Float16)b;
    return c.u;                       // lowers to v_cvt_pk_f16_f32
}

union HPack { unsigned int u[8]; v16h v; };

__launch_bounds__(256)
__global__ void obstacle_resblock_kernel(
    const float* __restrict__ traj, const float* __restrict__ speed,
    const float* __restrict__ obstacle,
    const float* __restrict__ W1, const float* __restrict__ b1,
    const float* __restrict__ W2, const float* __restrict__ b2,
    const float* __restrict__ Wo, const float* __restrict__ bo,
    float* __restrict__ out)
{
    const int tid  = threadIdx.x;
    const int lane = tid & 31;
    const int wave = tid >> 5;
    const int m    = lane & 15;   // sample column owned by this lane
    const int hi   = lane >> 4;   // half-wave

    const int tile = blockIdx.x * 8 + wave;
    const int ridx = tile * 16 + m;               // global row id

    // Arithmetic predicates (avoid exec-diamond if-conversion around loads):
    const float zsel = (hi == 0) ? 1.f : 0.f;     // layer-1 K-padding lanes
    const float msel = (m < 4)  ? 1.f : 0.f;      // layer-2 out-feature padding rows

    // ---- geometry (lane-local; lanes 16-31 duplicate rows 0-15's values) ----
    const int o   = ridx & (O_ - 1);
    const int tt  = (ridx >> 5) % T_;
    const int ped = ridx / (O_ * T_);
    const int bb  = ped >> 6;

    const float px = traj[(ped * T_ + tt) * 2 + 0];
    const float py = traj[(ped * T_ + tt) * 2 + 1];
    const float vx = speed[(ped * T_ + tt) * 2 + 0];
    const float vy = speed[(ped * T_ + tt) * 2 + 1];
    const float ox = obstacle[(bb * O_ + o) * 2 + 0];
    const float oy = obstacle[(bb * O_ + o) * 2 + 1];

    const float dx = ox - px, dy = oy - py;
    const float dist = sqrtf(dx * dx + dy * dy);
    const float inv  = 1.f / (dist + EPS_);
    const float dot  = vx * (dx * inv) + vy * (dy * inv);
    const float spd  = sqrtf(vx * vx + vy * vy) + EPS_;
    float cv = fminf(1.f, fmaxf(-1.f, dot / spd));
    const float ang = acosf(cv);
    const bool maskv = (dist <= SENSING_) && (ang <= HALF_ANGLE);

    // ---- B operand for layer 1: v4^T (4x16, K padded to 32).
    //      B elem e <-> K = 16*hi + e; only K<4 nonzero -> hi==0 lanes, dwords 0..1.
    HPack bv1;
    bv1.u[0] = pack2_(dx * zsel, dy * zsel);
    bv1.u[1] = pack2_(vx * zsel, vy * zsel);
    #pragma unroll
    for (int j = 2; j < 8; ++j) bv1.u[j] = 0u;

    // ---- layer 1 (transposed): h^T = W1^T @ v4^T + b1, 8 hidden tiles of 16 ----
    // D1: lane = sample column m, element e = hidden 16*t + 8*hi + e.
    unsigned int pk[8][4];
    #pragma unroll
    for (int t = 0; t < 8; ++t) {
        // A operand: W1^T tile. A elem e<8 <-> K=8*hi+e; only K<4 real -> hi==0 dwords 0..1.
        // All lanes load (in-bounds, coalesced); zeroing via multiply, not exec.
        float w0 = W1[0 * DIM_ + t * 16 + m] * zsel;
        float w1 = W1[1 * DIM_ + t * 16 + m] * zsel;
        float w2 = W1[2 * DIM_ + t * 16 + m] * zsel;
        float w3 = W1[3 * DIM_ + t * 16 + m] * zsel;
        HPack a1;
        a1.u[0] = pack2_(w0, w1);
        a1.u[1] = pack2_(w2, w3);
        #pragma unroll
        for (int j = 2; j < 8; ++j) a1.u[j] = 0u;

        v8f acc;
        #pragma unroll
        for (int e = 0; e < 8; ++e) acc[e] = b1[t * 16 + 8 * hi + e];
        acc = __builtin_amdgcn_wmma_f32_16x16x32_f16(false, a1.v, false, bv1.v,
                                                     (short)0, acc, false, false);
        #pragma unroll
        for (int j = 0; j < 4; ++j)
            pk[t][j] = pack2_(leaky_(acc[2 * j]), leaky_(acc[2 * j + 1]));
    }

    // ---- layer 2 (transposed): emb^T = W2^T @ h^T + b2, K=128 in 4 chunks ----
    v8f acc2;
    #pragma unroll
    for (int e = 0; e < 8; ++e) {
        int r = 8 * hi + e;
        acc2[e] = (r < 4) ? b2[r] : 0.f;
    }

    const int mc = m & 3;        // clamped output-feature index (keeps loads in-bounds)

    #pragma unroll
    for (int q = 0; q < 4; ++q) {
        // A operand: W2^T rows = out features (only m<4 real), K = hidden chunk.
        // A elem e<8 <-> k = 32q + 16*hi + e ; elem 8+e <-> k = 32q + 16*hi + 8 + e.
        HPack a2;
        #pragma unroll
        for (int j = 0; j < 4; ++j) {
            float wa = W2[(32 * q + 16 * hi + 2 * j + 0) * 4 + mc] * msel;
            float wb = W2[(32 * q + 16 * hi + 2 * j + 1) * 4 + mc] * msel;
            a2.u[j] = pack2_(wa, wb);
        }
        #pragma unroll
        for (int j = 0; j < 4; ++j) {
            float wa = W2[(32 * q + 16 * hi + 8 + 2 * j + 0) * 4 + mc] * msel;
            float wb = W2[(32 * q + 16 * hi + 8 + 2 * j + 1) * 4 + mc] * msel;
            a2.u[4 + j] = pack2_(wa, wb);
        }

        // B operand: h^T chunk (B elem e <-> k = 32q + 16*hi + e):
        //   hi==0: dwords 0..3 = own pk[2q],   dwords 4..7 = partner pk[2q]
        //   hi==1: dwords 0..3 = partner pk[2q+1], dwords 4..7 = own pk[2q+1]
        HPack bb2;
        #pragma unroll
        for (int j = 0; j < 4; ++j) {
            unsigned int send = hi ? pk[2 * q][j] : pk[2 * q + 1][j];
            unsigned int recv = (unsigned int)__shfl_xor((int)send, 16, 32);
            bb2.u[j]     = hi ? recv : pk[2 * q][j];
            bb2.u[4 + j] = hi ? pk[2 * q + 1][j] : recv;
        }
        acc2 = __builtin_amdgcn_wmma_f32_16x16x32_f16(false, a2.v, false, bb2.v,
                                                      (short)0, acc2, false, false);
    }

    // ---- epilogue: lanes 0-15 hold emb^T rows 0..3 for their own sample ----
    if (hi == 0) {
        float f0 = leaky_(acc2[0]) + dx;
        float f1 = leaky_(acc2[1]) + dy;
        float f2 = leaky_(acc2[2]) + vx;
        float f3 = leaky_(acc2[3]) + vy;
        float o0 = f0 * Wo[0] + f1 * Wo[2] + f2 * Wo[4] + f3 * Wo[6] + bo[0];
        float o1 = f0 * Wo[1] + f1 * Wo[3] + f2 * Wo[5] + f3 * Wo[7] + bo[1];
        o0 = 1.f / (1.f + __expf(-o0));
        o1 = 1.f / (1.f + __expf(-o1));
        out[ridx] = maskv ? 1.f : 0.f;               // mask block: N floats
        out[NROWS + 2 * ridx + 0] = o0;              // out block: 2N floats
        out[NROWS + 2 * ridx + 1] = o1;
    }
}

extern "C" void kernel_launch(void* const* d_in, const int* in_sizes, int n_in,
                              void* d_out, int out_size, void* d_ws, size_t ws_size,
                              hipStream_t stream) {
    const float* traj     = (const float*)d_in[0];
    const float* speed    = (const float*)d_in[1];
    const float* obstacle = (const float*)d_in[2];
    // d_in[3] = seq_start_end (int64): unused, segments are equal-length by construction
    const float* W1 = (const float*)d_in[4];
    const float* b1 = (const float*)d_in[5];
    const float* W2 = (const float*)d_in[6];
    const float* b2 = (const float*)d_in[7];
    const float* Wo = (const float*)d_in[8];
    const float* bo = (const float*)d_in[9];
    float* out = (float*)d_out;

    // 655360 rows / (16 rows/wave * 8 waves/block) = 5120 blocks
    const int blocks = NROWS / (16 * 8);
    obstacle_resblock_kernel<<<blocks, 256, 0, stream>>>(
        traj, speed, obstacle, W1, b1, W2, b2, Wo, bo, out);
}
